// QuantLinearAWQ_62861141344484
// MI455X (gfx1250) — compile-verified, matrix operand order checked by reading, not run on hardware
//
#include <hip/hip_runtime.h>
#include <stdint.h>

// ---------------------------------------------------------------------------
// AWQ 4-bit fused dequant + GEMM for gfx1250 (MI455X), bf16 WMMA, f32 accum.
// Double-buffered LDS software pipeline, magic-constant nibble dequant:
//   as_f32(0x43800000 | (nib<<16)) == 256 + 2*nib (exact), so
//   (w - z)*s == fma(256+2w, s/2, -(z+128)*s)  -> bfe + lshl_or + fma per weight.
// Block tile: 128(M) x 128(N), K-step 64. 256 threads = 8 waves, each wave
// owns a 32x64 sub-tile = 2x4 v_wmma_f32_16x16x32_bf16 accumulators.
// ---------------------------------------------------------------------------

typedef __attribute__((ext_vector_type(16))) __bf16 v16bf;
typedef __attribute__((ext_vector_type(8)))  float  v8f;

#define BM 128
#define BN 128
#define BK 64
#define PITCH 72   // row pitch in elements: 144 B, multiple of 16 B for b128 LDS ops

// Single-instruction f32x2 -> packed bf16x2 (lo = a, hi = b).
__device__ __forceinline__ uint32_t cvt_pk_bf16(float a, float b) {
    uint32_t r;
    asm("v_cvt_pk_bf16_f32 %0, %1, %2" : "=v"(r) : "v"(a), "v"(b));
    return r;
}

__global__ __launch_bounds__(256)
void awq_gemm_bf16_wmma(const float*    __restrict__ x,
                        const uint32_t* __restrict__ qweight,
                        const uint32_t* __restrict__ qzeros,
                        const float*    __restrict__ scales,
                        const float*    __restrict__ bias,
                        float*          __restrict__ out,
                        int M, int K, int N, int gshift)
{
    __shared__ __align__(16) uint16_t xs [2][BM][PITCH];  // x tile (bf16 bits), K contiguous
    __shared__ __align__(16) uint16_t wsT[2][BN][PITCH];  // W tile transposed [n][k]

    const int tid   = threadIdx.x;
    const int lane  = tid & 31;
    const int wave  = tid >> 5;
    const int waveM = wave & 3;      // 4 strips of 32 rows
    const int waveN = wave >> 2;     // 2 strips of 64 cols
    const int half  = lane >> 4;
    const int r16   = lane & 15;

    const int m0  = blockIdx.y * BM;
    const int n0  = blockIdx.x * BN;
    const int Nd8 = N >> 3;

    const int xl_c  = tid & 15;      // float4 column (16 per 64-k row)
    const int xl_r  = tid >> 4;      // base row (8 rows/thread, stride 16)
    const int dq_c  = tid & 15;      // packed int32 column
    const int dq_kp = tid >> 4;      // k-pair index

    // AWQ nibble reorder: logical col j uses shift 4*[0,4,1,5,2,6,3,7][j]
    const int rev_shift[8] = {0, 16, 4, 20, 8, 24, 12, 28};

    // Loop-invariant per-thread global pointers (loop-carried addressing is add-only)
    const float* xrow[8];
#pragma unroll
    for (int rr = 0; rr < 8; ++rr)
        xrow[rr] = x + (size_t)(m0 + xl_r + rr * 16) * K + 4 * xl_c;
    const uint32_t* qwp = qweight + (size_t)(2 * dq_kp) * Nd8 + (n0 >> 3) + dq_c;
    const uint32_t* qzp = qzeros + (n0 >> 3) + dq_c;
    const float*    scp = scales + n0 + 8 * dq_c;

    // Staged registers for the in-flight tile
    float4   xv[8];
    uint32_t qv[4];
    float    c0[8], c1[8];   // per-column dequant constants (change once per group)
    int      gcur = -1;

    auto fetch = [&](int k0) {          // issue global loads for tile at k0
#pragma unroll
        for (int rr = 0; rr < 8; ++rr)
            xv[rr] = *(const float4*)(xrow[rr] + k0);
        const size_t koff = (size_t)k0 * Nd8;
        qv[0] = qwp[koff];
        qv[1] = qwp[koff + (size_t)Nd8];
        qv[2] = qwp[koff + (size_t)32 * Nd8];
        qv[3] = qwp[koff + (size_t)33 * Nd8];
        const int g = k0 >> gshift;     // group is a power of two; BK divides group
        if (g != gcur) {                // uniform branch: reload zeros/scales per group only
            gcur = g;
            const uint32_t zv = qzp[(size_t)g * Nd8];
            const float4 sv0 = *(const float4*)(scp + (size_t)g * N);
            const float4 sv1 = *(const float4*)(scp + (size_t)g * N + 4);
            const float sj[8] = {sv0.x, sv0.y, sv0.z, sv0.w, sv1.x, sv1.y, sv1.z, sv1.w};
#pragma unroll
            for (int j = 0; j < 8; ++j) {
                const float zf = (float)((zv >> rev_shift[j]) & 15u);
                c1[j] = 0.5f * sj[j];
                c0[j] = -(zf + 128.0f) * sj[j];
            }
        }
    };

    auto stage = [&](int buf) {         // convert/dequant staged regs -> LDS buf
#pragma unroll
        for (int rr = 0; rr < 8; ++rr) {
            const int row = xl_r + rr * 16;
            uint2 p;
            p.x = cvt_pk_bf16(xv[rr].x, xv[rr].y);
            p.y = cvt_pk_bf16(xv[rr].z, xv[rr].w);
            *(uint2*)(&xs[buf][row][4 * xl_c]) = p;
        }
#pragma unroll
        for (int kb = 0; kb < 2; ++kb) {
            const int krow = kb * 32 + 2 * dq_kp;
            const uint32_t q0 = qv[kb * 2 + 0];
            const uint32_t q1 = qv[kb * 2 + 1];
#pragma unroll
            for (int j = 0; j < 8; ++j) {
                const int sh = rev_shift[j];
                const uint32_t t0 = (q0 >> sh) & 15u;        // v_bfe_u32
                const uint32_t t1 = (q1 >> sh) & 15u;
                const float f0 = __uint_as_float(0x43800000u | (t0 << 16)); // 256 + 2*t0
                const float f1 = __uint_as_float(0x43800000u | (t1 << 16));
                const float v0 = fmaf(f0, c1[j], c0[j]);     // (t0 - z) * s
                const float v1 = fmaf(f1, c1[j], c0[j]);
                // adjacent k pack into one dword store (wsT rows are k-contiguous)
                *(uint32_t*)(&wsT[buf][8 * dq_c + j][krow]) = cvt_pk_bf16(v0, v1);
            }
        }
    };

    v8f acc[2][4];
#pragma unroll
    for (int mi = 0; mi < 2; ++mi)
#pragma unroll
        for (int ni = 0; ni < 4; ++ni)
#pragma unroll
            for (int e = 0; e < 8; ++e) acc[mi][ni][e] = 0.0f;

    // Prologue: fill buffer 0
    fetch(0);
    stage(0);
    __syncthreads();

    const int nsteps = K / BK;
    for (int kt = 0; kt < nsteps; ++kt) {
        const int  buf     = kt & 1;
        const bool hasNext = (kt + 1 < nsteps);   // uniform branch: EXEC stays full

        if (hasNext) fetch((kt + 1) * BK);        // global loads overlap WMMA below
        if (kt + 2 < nsteps) {                    // 2-ahead L2 prefetch (global_prefetch_b8)
            __builtin_prefetch(xrow[0] + (kt + 2) * BK, 0, 1);
            __builtin_prefetch(qwp + (size_t)(kt + 2) * BK * Nd8, 0, 1);
        }

        // ------------------------------- WMMA -------------------------------
#pragma unroll
        for (int ks = 0; ks < 2; ++ks) {          // two K=32 sub-steps
            const int kbase = ks * 32 + half * 8; // lane's contiguous quads (ISA A/B layout)
            union { v16bf v; uint4 q[2]; } a[2], b[4];
#pragma unroll
            for (int mi = 0; mi < 2; ++mi) {
                const int row = waveM * 32 + mi * 16 + r16;
                a[mi].q[0] = *(const uint4*)(&xs[buf][row][kbase]);
                a[mi].q[1] = *(const uint4*)(&xs[buf][row][kbase + 16]);
            }
#pragma unroll
            for (int ni = 0; ni < 4; ++ni) {
                const int col = waveN * 64 + ni * 16 + r16;
                b[ni].q[0] = *(const uint4*)(&wsT[buf][col][kbase]);
                b[ni].q[1] = *(const uint4*)(&wsT[buf][col][kbase + 16]);
            }
#pragma unroll
            for (int mi = 0; mi < 2; ++mi)
#pragma unroll
                for (int ni = 0; ni < 4; ++ni)
                    acc[mi][ni] = __builtin_amdgcn_wmma_f32_16x16x32_bf16(
                        false, a[mi].v, false, b[ni].v,
                        (short)0, acc[mi][ni], false, false);
        }

        if (hasNext) stage(buf ^ 1);              // dequant VALU co-executes with WMMA pipe
        __syncthreads();                          // single barrier per K-step
    }

    // ------------------- store: C/D layout + bias -------------------
#pragma unroll
    for (int mi = 0; mi < 2; ++mi) {
#pragma unroll
        for (int ni = 0; ni < 4; ++ni) {
            const int n  = n0 + waveN * 64 + ni * 16 + r16;
            const float bv = bias[n];
#pragma unroll
            for (int rr = 0; rr < 8; ++rr) {
                const int m = m0 + waveM * 32 + mi * 16 + rr + 8 * half;
                out[(size_t)m * N + n] = acc[mi][ni][rr] + bv;
            }
        }
    }
}

extern "C" void kernel_launch(void* const* d_in, const int* in_sizes, int n_in,
                              void* d_out, int out_size, void* d_ws, size_t ws_size,
                              hipStream_t stream) {
    const float*    x       = (const float*)   d_in[0];
    const uint32_t* qweight = (const uint32_t*)d_in[1];
    const uint32_t* qzeros  = (const uint32_t*)d_in[2];
    const float*    scales  = (const float*)   d_in[3];
    const float*    bias    = (const float*)   d_in[4];
    float*          out     = (float*)d_out;

    const int N     = in_sizes[4];           // OUT = 11008
    const int Nd8   = N / 8;
    const int K     = in_sizes[1] / Nd8;     // IN  = 4096
    const int M     = in_sizes[0] / K;       // B*S = 8192
    const int NG    = in_sizes[3] / N;       // 32
    const int group = K / NG;                // 128 (power of two)
    const int gshift = __builtin_ctz(group);

    dim3 grid(N / BN, M / BM);               // 86 x 64
    awq_gemm_bf16_wmma<<<grid, 256, 0, stream>>>(x, qweight, qzeros, scales, bias, out,
                                                 M, K, N, gshift);
}
